// CAE_7447473291634
// MI455X (gfx1250) — matrix operand order repaired and checked by reading, not compiled, not run on hardware
//
#include <hip/hip_runtime.h>
#include <hip/hip_bf16.h>
#include <math.h>

typedef __attribute__((ext_vector_type(16))) _Float16 v16h;
typedef __attribute__((ext_vector_type(8)))  _Float16 v8h;
typedef __attribute__((ext_vector_type(2)))  _Float16 v2h;
typedef __attribute__((ext_vector_type(8)))  float    v8f;

#define PI_F 3.14159265358979323846f

__device__ __forceinline__ float dereg_f(float t, float mn, float mx) {
    return (t + 1.0f) * (mx - mn) * 0.5f + mn;
}

// ===========================================================================
// Staging kernels (f32 NCHW -> f16 NHWC channel-padded, weights, etc.)
// ===========================================================================

// x [B,C,HW] f32 -> Q [B,HW,CPAD] f16 (zero pad channels)
__global__ void nchw_to_nhwc_f16(const float* __restrict__ in, _Float16* __restrict__ out,
                                 int C, int CPAD, int HWpix, int total) {
    int id = blockIdx.x * blockDim.x + threadIdx.x;
    if (id >= total) return;
    int cp2 = CPAD >> 1;
    int c2  = id % cp2;
    int pix = id / cp2;             // pix in [0, B*HW)
    int b   = pix / HWpix;
    int rem = pix - b * HWpix;
    int c   = c2 * 2;
    int c0  = (c     < C) ? c     : (C - 1);
    int c1  = (c + 1 < C) ? c + 1 : (C - 1);
    float f0 = in[((size_t)b * C + c0) * HWpix + rem];
    float f1 = in[((size_t)b * C + c1) * HWpix + rem];
    v2h h;
    h.x = (_Float16)((c     < C) ? f0 : 0.0f);
    h.y = (_Float16)((c + 1 < C) ? f1 : 0.0f);
    ((v2h*)out)[(size_t)pix * cp2 + c2] = h;
}

// conv-out f32 NCHW -> relu? -> 2x2 pool -> f16 NHWC padded  [B,Ho,Wo,CPAD]
__global__ void pool_nchw_to_nhwc_f16(const float* __restrict__ in, _Float16* __restrict__ out,
                                      int C, int CPAD, int Hin, int do_relu, int total) {
    int id = blockIdx.x * blockDim.x + threadIdx.x;
    if (id >= total) return;
    int cp2 = CPAD >> 1;
    int Ho  = Hin >> 1;
    int c2  = id % cp2;
    int t   = id / cp2;
    int xo  = t % Ho;  t /= Ho;
    int yo  = t % Ho;  t /= Ho;
    int b   = t;
    int c   = 2 * c2;
    int c0  = (c     < C) ? c     : (C - 1);
    int c1  = (c + 1 < C) ? c + 1 : (C - 1);
    const float* p0 = in + (((size_t)b * C + c0) * Hin + 2 * yo) * Hin + 2 * xo;
    const float* p1 = in + (((size_t)b * C + c1) * Hin + 2 * yo) * Hin + 2 * xo;
    float m0 = fmaxf(fmaxf(p0[0], p0[1]), fmaxf(p0[Hin], p0[Hin + 1]));
    float m1 = fmaxf(fmaxf(p1[0], p1[1]), fmaxf(p1[Hin], p1[Hin + 1]));
    if (do_relu) { m0 = fmaxf(m0, 0.0f); m1 = fmaxf(m1, 0.0f); }
    v2h h;
    h.x = (_Float16)((c     < C) ? m0 : 0.0f);
    h.y = (_Float16)((c + 1 < C) ? m1 : 0.0f);
    int pixel = (b * Ho + yo) * Ho + xo;
    ((v2h*)out)[(size_t)pixel * cp2 + c2] = h;
}

// conv3 out f32 NCHW -> relu -> 2x2 pool -> f16 NCHW-flat [B,2048] (fc1 input)
__global__ void pool_flat_f16(const float* __restrict__ in, _Float16* __restrict__ out,
                              int C, int Hin, int total) {
    int id = blockIdx.x * blockDim.x + threadIdx.x;
    if (id >= total) return;
    int Ho = Hin >> 1;
    int xo = id % Ho;
    int t  = id / Ho;
    int yo = t % Ho;  t /= Ho;
    int ch = t % C;
    int b  = t / C;
    const float* ip = in + (((size_t)b * C + ch) * Hin + 2 * yo) * Hin + 2 * xo;
    float v = fmaxf(fmaxf(ip[0], ip[1]), fmaxf(ip[Hin], ip[Hin + 1]));
    out[id] = (_Float16)fmaxf(v, 0.0f);
}

// conv weights [COUT][CIN][3][3] f32 -> [9][COUT][CPAD] f16 (zero-pad channels)
__global__ void stage_wconv_f16(const float* __restrict__ w, _Float16* __restrict__ wq,
                                int CIN, int CPAD, int COUT, int total) {
    int id = blockIdx.x * blockDim.x + threadIdx.x;
    if (id >= total) return;
    int cp2  = CPAD >> 1;
    int c2   = id % cp2;
    int t    = id / cp2;
    int cout = t % COUT;
    int rs   = t / COUT;
    int c    = 2 * c2;
    int c0   = (c     < CIN) ? c     : (CIN - 1);
    int c1   = (c + 1 < CIN) ? c + 1 : (CIN - 1);
    float f0 = w[((size_t)cout * CIN + c0) * 9 + rs];
    float f1 = w[((size_t)cout * CIN + c1) * 9 + rs];
    v2h h;
    h.x = (_Float16)((c     < CIN) ? f0 : 0.0f);
    h.y = (_Float16)((c + 1 < CIN) ? f1 : 0.0f);
    ((v2h*)wq)[((size_t)rs * COUT + cout) * cp2 + c2] = h;
}

// fc weights [O][F] f32 -> [OT][F] f16 (zero rows beyond O)
__global__ void stage_wfc_f16(const float* __restrict__ w, _Float16* __restrict__ wq,
                              int O, int F, int total) {
    int id = blockIdx.x * blockDim.x + threadIdx.x;
    if (id >= total) return;
    int f2  = F >> 1;
    int c2  = id % f2;
    int row = id / f2;
    int c   = 2 * c2;
    int r0  = (row < O) ? row : 0;
    float f0 = w[(size_t)r0 * F + c];
    float f1 = w[(size_t)r0 * F + c + 1];
    v2h h;
    h.x = (_Float16)((row < O) ? f0 : 0.0f);
    h.y = (_Float16)((row < O) ? f1 : 0.0f);
    ((v2h*)wq)[(size_t)row * f2 + c2] = h;
}

// ===========================================================================
// conv3x3 SAME as 9 shifted GEMMs, K = CPAD (multiple of 32, no tail).
// xq: [B,H,W,CPAD] f16, wq: [9][COUT][CPAD] f16 staged through LDS.
// A frag: 2x ds_load_b128 ; B frag: 2x global_load_b128 + cndmask halo.
// One wave per 16cout x 16pix tile, 8 waves/block sharing the LDS A tile.
// ===========================================================================
template<int CPAD, int COUT, int H, int W>
__global__ void conv3x3_wmma(const _Float16* __restrict__ xq,
                             const _Float16* __restrict__ wq,
                             const float* __restrict__ bias,
                             float* __restrict__ out) {
    constexpr int KS = CPAD / 32;
    constexpr int HW = H * W;
    __shared__ _Float16 As[9 * KS * 16 * 32];

    const int tid      = threadIdx.x;
    const int lane     = tid & 31;
    const int wave     = tid >> 5;
    const int coutBase = blockIdx.y * 16;

    // cooperative weight-tile staging (dword granularity, fully coalesced)
    {
        const unsigned* src = (const unsigned*)wq;
        unsigned*       dst = (unsigned*)As;
        constexpr int ND = 9 * KS * 16 * 16;     // dwords
        #pragma unroll
        for (int d = tid; d < ND; d += 256) {
            int q  = d & 15;
            int m_ = (d >> 4) & 15;
            int t  = d >> 8;
            int ks = t % KS;
            int rs = t / KS;
            dst[d] = src[((size_t)rs * COUT + coutBase + m_) * (CPAD / 2) + ks * 16 + q];
        }
    }
    __syncthreads();

    const int m  = lane & 15;        // A row / B column / C column
    const int hh = lane >> 4;        // lane half
    const int pixTile = blockIdx.x * 8 + wave;
    const int pix = pixTile * 16 + m;
    const int bb  = pix / HW;
    const int rem = pix & (HW - 1);
    const int py  = rem / W;
    const int px  = rem & (W - 1);

    v8f c  = {};
    v8h z8 = {};
    #pragma unroll
    for (int r = 0; r < 3; ++r) {
        // prefetch next input row for this pixel column
        {
            int pr = (py + r < H) ? (py + r) : (H - 1);
            __builtin_prefetch((const void*)(xq + ((size_t)(bb * H + pr) * W + px) * CPAD), 0, 1);
        }
        #pragma unroll
        for (int s = 0; s < 3; ++s) {
            const int rs = r * 3 + s;
            const int yy = py + r - 1, xx = px + s - 1;
            const bool valid = ((unsigned)yy < (unsigned)H) && ((unsigned)xx < (unsigned)W);
            const int yyc = valid ? yy : 0;
            const int xxc = valid ? xx : 0;
            const _Float16* bp = xq + ((size_t)(bb * H + yyc) * W + xxc) * CPAD;
            #pragma unroll
            for (int ks = 0; ks < KS; ++ks) {
                const _Float16* arow = &As[((rs * KS + ks) * 16 + m) * 32];
                v8h alo = *(const v8h*)(arow + hh * 8);        // K = 32ks+8h .. +7
                v8h ahi = *(const v8h*)(arow + 16 + hh * 8);   // K = 32ks+16+8h .. +7
                v8h b0  = *(const v8h*)(bp + ks * 32 + hh * 8);       // K = 32ks+8h .. +7
                v8h b1  = *(const v8h*)(bp + ks * 32 + 16 + hh * 8);  // K = 32ks+16+8h .. +7
                b0 = valid ? b0 : z8;
                b1 = valid ? b1 : z8;
                v16h a = __builtin_shufflevector(alo, ahi, 0,1,2,3,4,5,6,7,8,9,10,11,12,13,14,15);
                v16h b = __builtin_shufflevector(b0,  b1,  0,1,2,3,4,5,6,7,8,9,10,11,12,13,14,15);
                c = __builtin_amdgcn_wmma_f32_16x16x32_f16(false, a, false, b,
                                                           (short)0, c, false, false);
            }
        }
    }
    // C/D layout: lane(n,hh), VGPR r8 -> M(channel) = r8 + 8*hh
    #pragma unroll
    for (int r8 = 0; r8 < 8; ++r8) {
        int ch = coutBase + r8 + 8 * hh;
        out[(((size_t)bb * COUT + ch) * H + py) * W + px] = c[r8] + bias[ch];
    }
}

// ===========================================================================
// FC layer via WMMA on staged f16 operands. D[o-tile 16 x batch-tile 16].
// MODE 0: relu -> f16 activations.  MODE 1: clamp [-1,1] -> f32 p[B][6].
// ===========================================================================
template<int OREAL, int F, int MODE, typename OUT_T>
__global__ void fc_wmma(const _Float16* __restrict__ h, const _Float16* __restrict__ wq,
                        const float* __restrict__ bias, OUT_T* __restrict__ out) {
    const int lane  = threadIdx.x & 31;
    const int bBase = blockIdx.x * 16;
    const int oBase = blockIdx.y * 16;
    const int m  = lane & 15;
    const int hh = lane >> 4;
    const int bn = bBase + m;
    const _Float16* ap = wq + (size_t)(oBase + m) * F;
    const _Float16* bp = h  + (size_t)bn * F;

    v8f c = {};
    #pragma unroll 4
    for (int ks = 0; ks < F / 32; ++ks) {
        v8h alo = *(const v8h*)(ap + ks * 32 + hh * 8);
        v8h ahi = *(const v8h*)(ap + ks * 32 + 16 + hh * 8);
        v8h b0  = *(const v8h*)(bp + ks * 32 + hh * 8);
        v8h b1  = *(const v8h*)(bp + ks * 32 + 16 + hh * 8);
        v16h a = __builtin_shufflevector(alo, ahi, 0,1,2,3,4,5,6,7,8,9,10,11,12,13,14,15);
        v16h b = __builtin_shufflevector(b0,  b1,  0,1,2,3,4,5,6,7,8,9,10,11,12,13,14,15);
        c = __builtin_amdgcn_wmma_f32_16x16x32_f16(false, a, false, b,
                                                   (short)0, c, false, false);
    }
    #pragma unroll
    for (int r8 = 0; r8 < 8; ++r8) {
        int o = oBase + r8 + 8 * hh;
        if (o < OREAL) {
            float v = c[r8] + bias[o];
            if (MODE == 0) v = fmaxf(v, 0.0f);
            else           v = fminf(fmaxf(v, -1.0f), 1.0f);
            out[(size_t)bn * OREAL + o] = (OUT_T)v;
        }
    }
}

// ===========================================================================
// physics decoder (unchanged: bandwidth / transcendental bound)
// ===========================================================================
__global__ void init_minmax(unsigned* mm) {
    if (threadIdx.x == 0) { mm[0] = 0x7f800000u; mm[1] = 0u; }
}

__global__ void decoder_fields(const float* __restrict__ p,
                               const float* __restrict__ xxg,
                               const float* __restrict__ yyg,
                               float* __restrict__ sbraw,
                               float* __restrict__ vout,
                               unsigned* __restrict__ mm,
                               const int* __restrict__ dvp) {
    int id  = blockIdx.x * blockDim.x + threadIdx.x;   // < 64*4096
    int b   = id >> 12;
    int pix = id & 4095;
    const float* pb = p + b * 6;
    float p0 = pb[0], p1 = pb[1], p2 = pb[2], p3 = pb[3], p4 = pb[4], p5 = pb[5];

    float pos = atan2f(p0, p1) + PI_F;
    float inc = dereg_f(p2, 5.0f, PI_F * 0.5f);
    float a   = dereg_f(p3, 0.1f, 0.5f)  * 32.0f;   // * shape/2
    float ah  = dereg_f(p4, 0.01f, 0.1f) * 32.0f;
    float Vh  = dereg_f(p5, 50.0f, 500.0f);

    float X = xxg[pix], Y = yyg[pix];
    float sp = sinf(pos), cp = cosf(pos);
    float xt = -X * sp + Y * cp;
    float yt = (X * cp + Y * sp) / sinf(PI_F * 0.5f - inc);
    float rr = sqrtf(xt * xt + yt * yt);

    float sb = expf(-rr / a);
    sbraw[id] = sb;
    atomicMin(&mm[0], __float_as_uint(sb));   // sb > 0 -> bit order == value order
    atomicMax(&mm[1], __float_as_uint(sb));

    float vel = sqrtf(Vh * Vh * (1.0f - ah / rr * atanf(rr / ah)));
    vel *= cosf(atan2f(yt, xt)) * sinf(inc);
    float dv   = (float)(*dvp);
    float vlim = 120.0f * dv * 0.5f;
    vel = (vel < -vlim || vel > vlim) ? 0.0f : vel;
    vout[id] = floorf(vel / dv);
}

__global__ void decoder_sbnorm(const float* __restrict__ sbraw,
                               const unsigned* __restrict__ mm,
                               float* __restrict__ sbout) {
    int id = blockIdx.x * blockDim.x + threadIdx.x;
    float mn = __uint_as_float(mm[0]);
    float mx = __uint_as_float(mm[1]);
    sbout[id] = (sbraw[id] - mn) / (mx - mn);
}

__global__ void decoder_binning(const float* __restrict__ sbout,
                                const float* __restrict__ vout,
                                float* __restrict__ cube) {
    int id  = blockIdx.x * blockDim.x + threadIdx.x;  // < 64*4096
    int b   = id >> 12;
    int pix = id & 4095;
    float amp = sbout[id];
    float mu  = vout[id] + 60.0f;                      // + width/2
    const float a_norm = 0.13298076013381091f;         // 1/(3*sqrt(2*pi))
    float* base = cube + (size_t)b * 120 * 4096 + pix;
    #pragma unroll 4
    for (int z = 0; z < 120; ++z) {
        float d = ((float)z - mu) * (1.0f / 3.0f);
        base[(size_t)z * 4096] = amp * a_norm * expf(-0.5f * d * d);
    }
}

// ===========================================================================
extern "C" void kernel_launch(void* const* d_in, const int* in_sizes, int n_in,
                              void* d_out, int out_size, void* d_ws, size_t ws_size,
                              hipStream_t stream) {
    (void)in_sizes; (void)n_in; (void)out_size; (void)ws_size;

    const float* x   = (const float*)d_in[0];
    const float* xxg = (const float*)d_in[1];
    const float* yyg = (const float*)d_in[2];
    const float* w0  = (const float*)d_in[5];
    const float* b0  = (const float*)d_in[6];
    const float* w1  = (const float*)d_in[7];
    const float* b1  = (const float*)d_in[8];
    const float* w2  = (const float*)d_in[9];
    const float* b2  = (const float*)d_in[10];
    const float* w3  = (const float*)d_in[11];
    const float* b3  = (const float*)d_in[12];
    const float* lw1 = (const float*)d_in[13];
    const float* lb1 = (const float*)d_in[14];
    const float* lw2 = (const float*)d_in[15];
    const float* lb2 = (const float*)d_in[16];
    const float* lw3 = (const float*)d_in[17];
    const float* lb3 = (const float*)d_in[18];
    const int*   dv  = (const int*)d_in[19];

    float* out = (float*)d_out;
    const size_t CUBE_N = 64ull * 120 * 4096;    // 31,457,280
    const size_t V_OFF  = CUBE_N;
    const size_t SB_OFF = V_OFF + 64ull * 4096;  // 31,719,424
    float* v_out  = out + V_OFF;
    float* sb_out = out + SB_OFF;

    // ---- scratch layout (~90 MB) ----
    float*     P0  = (float*)d_ws;              // conv f32 out, max 4,194,304 f
    _Float16*  Q   = (_Float16*)(P0 + 4194304); // NHWC f16 stage, max 33,554,432 h
    _Float16*  WQ  = Q + 33554432;              // staged conv weights, max 73,728 h
    _Float16*  H1  = WQ + 73728;                // fc1 input  [64][2048] f16
    _Float16*  A1  = H1 + 131072;               // fc1 output [64][1024] f16
    _Float16*  A2  = A1 + 65536;                // fc2 output [64][256]  f16
    _Float16*  LWQ = A2 + 16384;                // staged fc weights, max 2,097,152 h
    float*     Pp  = (float*)(LWQ + 2097152);   // p [64][6] f32 (padded region)
    float*     SB  = Pp + 512;                  // raw sb [64][4096] f32
    unsigned*  MM  = (unsigned*)(SB + 262144);  // global min/max bits

    // ---- encoder ----
    nchw_to_nhwc_f16<<<65536, 256, 0, stream>>>(x, Q, 120, 128, 4096, 16777216);
    stage_wconv_f16<<<(9216 + 255) / 256, 256, 0, stream>>>(w0, WQ, 120, 128, 16, 9216);
    conv3x3_wmma<128, 16, 64, 64><<<dim3(2048, 1), 256, 0, stream>>>(Q, WQ, b0, P0);

    pool_nchw_to_nhwc_f16<<<(1048576 + 255) / 256, 256, 0, stream>>>(P0, Q, 16, 32, 64, 0, 1048576);
    stage_wconv_f16<<<(4608 + 255) / 256, 256, 0, stream>>>(w1, WQ, 16, 32, 32, 4608);
    conv3x3_wmma<32, 32, 32, 32><<<dim3(512, 2), 256, 0, stream>>>(Q, WQ, b1, P0);

    pool_nchw_to_nhwc_f16<<<(262144 + 255) / 256, 256, 0, stream>>>(P0, Q, 32, 32, 32, 1, 262144);
    stage_wconv_f16<<<(9216 + 255) / 256, 256, 0, stream>>>(w2, WQ, 32, 32, 64, 9216);
    conv3x3_wmma<32, 64, 16, 16><<<dim3(128, 4), 256, 0, stream>>>(Q, WQ, b2, P0);

    pool_nchw_to_nhwc_f16<<<(131072 + 255) / 256, 256, 0, stream>>>(P0, Q, 64, 64, 16, 1, 131072);
    stage_wconv_f16<<<(36864 + 255) / 256, 256, 0, stream>>>(w3, WQ, 64, 64, 128, 36864);
    conv3x3_wmma<64, 128, 8, 8><<<dim3(32, 8), 256, 0, stream>>>(Q, WQ, b3, P0);

    pool_flat_f16<<<(131072 + 255) / 256, 256, 0, stream>>>(P0, H1, 128, 8, 131072);

    // ---- FC head ----
    stage_wfc_f16<<<(1048576 + 255) / 256, 256, 0, stream>>>(lw1, LWQ, 1024, 2048, 1048576);
    fc_wmma<1024, 2048, 0, _Float16><<<dim3(4, 64), 32, 0, stream>>>(H1, LWQ, lb1, A1);

    stage_wfc_f16<<<(131072 + 255) / 256, 256, 0, stream>>>(lw2, LWQ, 256, 1024, 131072);
    fc_wmma<256, 1024, 0, _Float16><<<dim3(4, 16), 32, 0, stream>>>(A1, LWQ, lb2, A2);

    stage_wfc_f16<<<(2048 + 255) / 256, 256, 0, stream>>>(lw3, LWQ, 6, 256, 2048);
    fc_wmma<6, 256, 1, float><<<dim3(4, 1), 32, 0, stream>>>(A2, LWQ, lb3, Pp);

    // ---- physics decoder ----
    init_minmax<<<1, 32, 0, stream>>>(MM);
    decoder_fields <<<1024, 256, 0, stream>>>(Pp, xxg, yyg, SB, v_out, MM, dv);
    decoder_sbnorm <<<1024, 256, 0, stream>>>(SB, MM, sb_out);
    decoder_binning<<<1024, 256, 0, stream>>>(sb_out, v_out, out);
}
